// MoE_40570261078622
// MI455X (gfx1250) — compile-verified, matrix operand order checked by reading, not run on hardware
//
#include <hip/hip_runtime.h>
#include <hip/hip_bf16.h>

// MoE top-2 SwiGLU for MI455X (gfx1250): bf16 WMMA, fp32 accum, double-buffered LDS.

typedef __bf16  bf16_t;
typedef bf16_t  v16bf __attribute__((ext_vector_type(16)));
typedef float   v8f   __attribute__((ext_vector_type(8)));

static constexpr int T = 8192;   // B*S tokens
static constexpr int H = 896;    // hidden
static constexpr int I = 4864;   // intermediate
static constexpr int E = 4;      // experts
static constexpr int LDT = 40;   // LDS tile row stride (bf16 elems); 80B -> conflict-free b128 reads

// ---------- fp32 -> bf16 (round-to-nearest-even) ----------
__device__ __forceinline__ unsigned short f2bf(float f) {
    unsigned u = __float_as_uint(f);
    unsigned r = u + 0x7FFFu + ((u >> 16) & 1u);
    return (unsigned short)(r >> 16);
}
__device__ __forceinline__ unsigned int pack2(float a, float b) {
    return (unsigned int)f2bf(a) | ((unsigned int)f2bf(b) << 16);
}

__global__ __launch_bounds__(256) void cvt_f32_bf16(const float* __restrict__ src,
                                                    unsigned short* __restrict__ dst,
                                                    int n8) {
    int i = blockIdx.x * 256 + threadIdx.x;
    if (i >= n8) return;
    const float4* s = (const float4*)src + (size_t)i * 2;
    float4 f0 = s[0], f1 = s[1];
    uint4 o;
    o.x = pack2(f0.x, f0.y);
    o.y = pack2(f0.z, f0.w);
    o.z = pack2(f1.x, f1.y);
    o.w = pack2(f1.z, f1.w);
    ((uint4*)dst)[i] = o;
}

// ---------- router: logits -> softmax(f32) -> top2 -> renorm -> per-expert lists ----------
__global__ __launch_bounds__(128) void moe_router(const float* __restrict__ x,
                                                  const float* __restrict__ gate_w,
                                                  int* __restrict__ counts,
                                                  int* __restrict__ tok,
                                                  float* __restrict__ slotw) {
    __shared__ float gw[E * H];
    int tid = threadIdx.x;
    for (int i = tid; i < E * H; i += 128) gw[i] = gate_w[i];
    __syncthreads();

    int t = blockIdx.x * 128 + tid;
    const float4* xr = (const float4*)(x + (size_t)t * H);
    float a0 = 0.f, a1 = 0.f, a2 = 0.f, a3 = 0.f;
    for (int h4 = 0; h4 < H / 4; ++h4) {
        float4 f = xr[h4];
        int h = h4 * 4;
        a0 += f.x * gw[0 * H + h] + f.y * gw[0 * H + h + 1] + f.z * gw[0 * H + h + 2] + f.w * gw[0 * H + h + 3];
        a1 += f.x * gw[1 * H + h] + f.y * gw[1 * H + h + 1] + f.z * gw[1 * H + h + 2] + f.w * gw[1 * H + h + 3];
        a2 += f.x * gw[2 * H + h] + f.y * gw[2 * H + h + 1] + f.z * gw[2 * H + h + 2] + f.w * gw[2 * H + h + 3];
        a3 += f.x * gw[3 * H + h] + f.y * gw[3 * H + h + 1] + f.z * gw[3 * H + h + 2] + f.w * gw[3 * H + h + 3];
    }
    float l[4] = {a0, a1, a2, a3};
    float m = fmaxf(fmaxf(l[0], l[1]), fmaxf(l[2], l[3]));
    float p[4]; float s = 0.f;
    #pragma unroll
    for (int e = 0; e < 4; ++e) { p[e] = __expf(l[e] - m); s += p[e]; }
    #pragma unroll
    for (int e = 0; e < 4; ++e) p[e] /= s;

    int b0 = 0;
    #pragma unroll
    for (int e = 1; e < 4; ++e) if (p[e] > p[b0]) b0 = e;       // lowest index wins ties
    int b1 = (b0 == 0) ? 1 : 0;
    #pragma unroll
    for (int e = 0; e < 4; ++e) if (e != b0 && p[e] > p[b1]) b1 = e;

    float s2 = p[b0] + p[b1];
    int s0 = atomicAdd(&counts[b0], 1);
    tok[b0 * T + s0] = t;  slotw[b0 * T + s0] = p[b0] / s2;
    int s1 = atomicAdd(&counts[b1], 1);
    tok[b1 * T + s1] = t;  slotw[b1 * T + s1] = p[b1] / s2;
}

// ---------- WMMA fragment load: two b128 LDS reads per ISA 16-bit A/B layout ----------
__device__ __forceinline__ v16bf load_frag(const unsigned short* p) {
    union { v16bf v; uint4 u[2]; } f;
    f.u[0] = *(const uint4*)(p);
    f.u[1] = *(const uint4*)(p + 16);
    return f.v;
}

// ---------- GEMM1: act[row, :] = silu(x[tok[row]]·Wg^T) * (x[tok[row]]·Wu^T)  (bf16 out) ----------
// BM=128, BN=64, BK=32; 8 waves, each owns 32x32 with dual (G,U) accumulators.
// Double-buffered LDS: one barrier per K-step, global loads overlap WMMA.
__global__ __launch_bounds__(256) void moe_gemm1(const unsigned short* __restrict__ xbf,
                                                 const unsigned short* __restrict__ Wg,
                                                 const unsigned short* __restrict__ Wu,
                                                 unsigned short* __restrict__ act,
                                                 const int* __restrict__ tok,
                                                 const int* __restrict__ count_ptr) {
    const int cnt = *count_ptr;
    const int rowBase = blockIdx.y * 128;
    if (rowBase >= cnt) return;                 // true top-2 work: skip empty M-blocks
    const int colBase = blockIdx.x * 64;

    __shared__ unsigned short As[2][128 * LDT];
    __shared__ unsigned short Bgs[2][64 * LDT];
    __shared__ unsigned short Bus[2][64 * LDT];
    __shared__ int tokS[128];

    const int tid = threadIdx.x;
    if (tid < 128) tokS[tid] = (rowBase + tid < cnt) ? tok[rowBase + tid] : 0;
    __syncthreads();

    // fixed per-thread staging assignment (hoisted out of K loop)
    const int rA0 = tid >> 2, rA1 = rA0 + 64;       // A rows (128 rows, 2 chunks/thread)
    const int kc  = (tid & 3) * 8;                   // 16B chunk offset within 32-wide K slice
    const unsigned short* pA0 = xbf + (size_t)tokS[rA0] * H + kc;
    const unsigned short* pA1 = xbf + (size_t)tokS[rA1] * H + kc;
    const unsigned short* pBg = Wg + (size_t)(colBase + rA0) * H + kc;   // B rows 0..63
    const unsigned short* pBu = Wu + (size_t)(colBase + rA0) * H + kc;

    const int lane = tid & 31, wave = tid >> 5;
    const int wm = (wave & 3) * 32;
    const int wn = (wave >> 2) * 32;
    const int fr = lane & 15;           // fragment row within 16x16 tile
    const int hk = (lane >> 4) * 8;     // K sub-offset per ISA layout

    v8f accG[2][2], accU[2][2];
    #pragma unroll
    for (int mi = 0; mi < 2; ++mi)
        #pragma unroll
        for (int ni = 0; ni < 2; ++ni) { accG[mi][ni] = (v8f)0.f; accU[mi][ni] = (v8f)0.f; }

    // preload tile 0
    uint4 rgA0 = *(const uint4*)(pA0);
    uint4 rgA1 = *(const uint4*)(pA1);
    uint4 rgBg = *(const uint4*)(pBg);
    uint4 rgBu = *(const uint4*)(pBu);
    *(uint4*)&As[0][rA0 * LDT + kc] = rgA0;
    *(uint4*)&As[0][rA1 * LDT + kc] = rgA1;
    *(uint4*)&Bgs[0][rA0 * LDT + kc] = rgBg;
    *(uint4*)&Bus[0][rA0 * LDT + kc] = rgBu;
    __syncthreads();

    constexpr int KT = H / 32;   // 28
    #pragma unroll 2
    for (int kk = 0; kk < KT; ++kk) {
        const int cur = kk & 1, nxt = cur ^ 1;
        if (kk + 1 < KT) {                       // issue next tile's global loads early
            const int k0 = (kk + 1) * 32;
            rgA0 = *(const uint4*)(pA0 + k0);
            rgA1 = *(const uint4*)(pA1 + k0);
            rgBg = *(const uint4*)(pBg + k0);
            rgBu = *(const uint4*)(pBu + k0);
        }

        v16bf a[2];
        #pragma unroll
        for (int mi = 0; mi < 2; ++mi)
            a[mi] = load_frag(&As[cur][(wm + mi * 16 + fr) * LDT + hk]);
        #pragma unroll
        for (int ni = 0; ni < 2; ++ni) {
            v16bf bg = load_frag(&Bgs[cur][(wn + ni * 16 + fr) * LDT + hk]);
            #pragma unroll
            for (int mi = 0; mi < 2; ++mi)
                accG[mi][ni] = __builtin_amdgcn_wmma_f32_16x16x32_bf16(
                    false, a[mi], false, bg, (short)0, accG[mi][ni], false, false);
            v16bf bu = load_frag(&Bus[cur][(wn + ni * 16 + fr) * LDT + hk]);
            #pragma unroll
            for (int mi = 0; mi < 2; ++mi)
                accU[mi][ni] = __builtin_amdgcn_wmma_f32_16x16x32_bf16(
                    false, a[mi], false, bu, (short)0, accU[mi][ni], false, false);
        }

        if (kk + 1 < KT) {                       // fill other buffer; single barrier/iter
            *(uint4*)&As[nxt][rA0 * LDT + kc] = rgA0;
            *(uint4*)&As[nxt][rA1 * LDT + kc] = rgA1;
            *(uint4*)&Bgs[nxt][rA0 * LDT + kc] = rgBg;
            *(uint4*)&Bus[nxt][rA0 * LDT + kc] = rgBu;
            __syncthreads();
        }
    }

    const int laneHiM = (lane >> 4) * 8;
    const int nIn = lane & 15;
    #pragma unroll
    for (int mi = 0; mi < 2; ++mi)
        #pragma unroll
        for (int ni = 0; ni < 2; ++ni)
            #pragma unroll
            for (int r = 0; r < 8; ++r) {
                int M = rowBase + wm + mi * 16 + laneHiM + r;
                int N = colBase + wn + ni * 16 + nIn;
                float g = accG[mi][ni][r];
                float u = accU[mi][ni][r];
                // silu via fast v_rcp (bf16 output -> ~1ulp rcp is plenty)
                float sg = g * __builtin_amdgcn_rcpf(1.f + __expf(-g));
                act[(size_t)M * I + N] = f2bf(sg * u);
            }
}

// ---------- GEMM2: out[tok[row], :] += w[row] * (act[row,:]·Wd^T) ----------
// BM=128, BN=128, BK=32; 8 waves, each owns 32x64. Double-buffered LDS.
__global__ __launch_bounds__(256) void moe_gemm2(const unsigned short* __restrict__ act,
                                                 const unsigned short* __restrict__ Wd,
                                                 float* __restrict__ out,
                                                 const int* __restrict__ tok,
                                                 const float* __restrict__ slotw,
                                                 const int* __restrict__ count_ptr) {
    const int cnt = *count_ptr;
    const int rowBase = blockIdx.y * 128;
    if (rowBase >= cnt) return;
    const int colBase = blockIdx.x * 128;

    __shared__ unsigned short As[2][128 * LDT];
    __shared__ unsigned short Bs[2][128 * LDT];
    __shared__ int   tokS[128];
    __shared__ float wS[128];

    const int tid = threadIdx.x;
    if (tid < 128) {
        bool v = rowBase + tid < cnt;
        tokS[tid] = v ? tok[rowBase + tid] : 0;
        wS[tid]   = v ? slotw[rowBase + tid] : 0.f;
    }
    __syncthreads();

    const int rA0 = tid >> 2, rA1 = rA0 + 64;
    const int kc  = (tid & 3) * 8;
    const unsigned short* pA0 = act + (size_t)(rowBase + rA0) * I + kc;
    const unsigned short* pA1 = act + (size_t)(rowBase + rA1) * I + kc;
    const unsigned short* pB0 = Wd + (size_t)(colBase + rA0) * I + kc;
    const unsigned short* pB1 = Wd + (size_t)(colBase + rA1) * I + kc;

    const int lane = tid & 31, wave = tid >> 5;
    const int wm = (wave & 3) * 32;
    const int wn = (wave >> 2) * 64;
    const int fr = lane & 15;
    const int hk = (lane >> 4) * 8;

    v8f acc[2][4];
    #pragma unroll
    for (int mi = 0; mi < 2; ++mi)
        #pragma unroll
        for (int ni = 0; ni < 4; ++ni) acc[mi][ni] = (v8f)0.f;

    uint4 rgA0 = *(const uint4*)(pA0);
    uint4 rgA1 = *(const uint4*)(pA1);
    uint4 rgB0 = *(const uint4*)(pB0);
    uint4 rgB1 = *(const uint4*)(pB1);
    *(uint4*)&As[0][rA0 * LDT + kc] = rgA0;
    *(uint4*)&As[0][rA1 * LDT + kc] = rgA1;
    *(uint4*)&Bs[0][rA0 * LDT + kc] = rgB0;
    *(uint4*)&Bs[0][rA1 * LDT + kc] = rgB1;
    __syncthreads();

    constexpr int KT = I / 32;   // 152
    #pragma unroll 2
    for (int kk = 0; kk < KT; ++kk) {
        const int cur = kk & 1, nxt = cur ^ 1;
        if (kk + 1 < KT) {
            const int k0 = (kk + 1) * 32;
            rgA0 = *(const uint4*)(pA0 + k0);
            rgA1 = *(const uint4*)(pA1 + k0);
            rgB0 = *(const uint4*)(pB0 + k0);
            rgB1 = *(const uint4*)(pB1 + k0);
        }

        v16bf a[2];
        #pragma unroll
        for (int mi = 0; mi < 2; ++mi)
            a[mi] = load_frag(&As[cur][(wm + mi * 16 + fr) * LDT + hk]);
        #pragma unroll
        for (int ni = 0; ni < 4; ++ni) {
            v16bf b = load_frag(&Bs[cur][(wn + ni * 16 + fr) * LDT + hk]);
            #pragma unroll
            for (int mi = 0; mi < 2; ++mi)
                acc[mi][ni] = __builtin_amdgcn_wmma_f32_16x16x32_bf16(
                    false, a[mi], false, b, (short)0, acc[mi][ni], false, false);
        }

        if (kk + 1 < KT) {
            *(uint4*)&As[nxt][rA0 * LDT + kc] = rgA0;
            *(uint4*)&As[nxt][rA1 * LDT + kc] = rgA1;
            *(uint4*)&Bs[nxt][rA0 * LDT + kc] = rgB0;
            *(uint4*)&Bs[nxt][rA1 * LDT + kc] = rgB1;
            __syncthreads();
        }
    }

    const int laneHiM = (lane >> 4) * 8;
    const int nIn = lane & 15;
    #pragma unroll
    for (int mi = 0; mi < 2; ++mi)
        #pragma unroll
        for (int ni = 0; ni < 4; ++ni)
            #pragma unroll
            for (int r = 0; r < 8; ++r) {
                int ml = wm + mi * 16 + laneHiM + r;
                if (rowBase + ml < cnt) {            // garbage rows are never scattered
                    int token = tokS[ml];
                    float wgt = wS[ml];
                    int N = colBase + wn + ni * 16 + nIn;
                    out[(size_t)token * H + N] += wgt * acc[mi][ni][r];
                }
            }
}

// ---------- host ----------
extern "C" void kernel_launch(void* const* d_in, const int* in_sizes, int n_in,
                              void* d_out, int out_size, void* d_ws, size_t ws_size,
                              hipStream_t stream) {
    (void)in_sizes; (void)n_in; (void)ws_size;
    const float* x      = (const float*)d_in[0];   // [4,2048,896]
    const float* gate_w = (const float*)d_in[1];   // [4,896]
    const float* Wg     = (const float*)d_in[2];   // [4,4864,896]
    const float* Wu     = (const float*)d_in[3];   // [4,4864,896]
    const float* Wd     = (const float*)d_in[4];   // [4,896,4864]
    float* out = (float*)d_out;                    // [4,2048,896]

    // workspace layout (~200 MB total)
    char* ws = (char*)d_ws;
    size_t off = 0;
    auto alloc = [&](size_t bytes) -> void* {
        void* p = ws + off;
        off += (bytes + 255) & ~(size_t)255;
        return p;
    };
    unsigned short* xbf   = (unsigned short*)alloc((size_t)T * H * 2);
    unsigned short* Wgbf  = (unsigned short*)alloc((size_t)E * I * H * 2);
    unsigned short* Wubf  = (unsigned short*)alloc((size_t)E * I * H * 2);
    unsigned short* Wdbf  = (unsigned short*)alloc((size_t)E * H * I * 2);
    unsigned short* actb  = (unsigned short*)alloc((size_t)T * I * 2);
    int*   counts = (int*)alloc(256);
    int*   tok    = (int*)alloc((size_t)E * T * 4);
    float* slotw  = (float*)alloc((size_t)E * T * 4);

    (void)hipMemsetAsync(out, 0, (size_t)out_size * sizeof(float), stream);
    (void)hipMemsetAsync(counts, 0, E * sizeof(int), stream);

    {   // bf16 conversions
        int n8 = T * H / 8;
        cvt_f32_bf16<<<(n8 + 255) / 256, 256, 0, stream>>>(x, xbf, n8);
        int w8 = E * I * H / 8;
        cvt_f32_bf16<<<(w8 + 255) / 256, 256, 0, stream>>>(Wg, Wgbf, w8);
        cvt_f32_bf16<<<(w8 + 255) / 256, 256, 0, stream>>>(Wu, Wubf, w8);
        cvt_f32_bf16<<<(w8 + 255) / 256, 256, 0, stream>>>(Wd, Wdbf, w8);
    }

    moe_router<<<T / 128, 128, 0, stream>>>(x, gate_w, counts, tok, slotw);

    for (int e = 0; e < E; ++e) {
        moe_gemm1<<<dim3(I / 64, T / 128), 256, 0, stream>>>(
            xbf, Wgbf + (size_t)e * I * H, Wubf + (size_t)e * I * H,
            actb, tok + (size_t)e * T, counts + e);
        moe_gemm2<<<dim3(H / 128, T / 128), 256, 0, stream>>>(
            actb, Wdbf + (size_t)e * H * I, out,
            tok + (size_t)e * T, slotw + (size_t)e * T, counts + e);
    }
}
